// GATConv_57629871177996
// MI455X (gfx1250) — matrix operand order repaired and checked by reading, not compiled, not run on hardware
//
#include <hip/hip_runtime.h>

// GAT dense attention, fused flash-style for MI455X (gfx1250, wave32, WMMA).
//
// N=4096, F=256, H=4, C=64, H*C=256.
// Dominant compute: out = softmax(logits) @ xp -> 8.6 GFLOP, v_wmma_f32_16x16x32_f16.
// Dominant bytes: adjacency a (67 MB, ~L2 resident, 23.3 TB/s HBM behind it).
// Logits are rank-1 (s_i + n_j): recomputed on the fly, never materialized.
// Both xp and W are stored in WMMA B-fragment order so every GEMM inner loop
// feeds WMMA straight from b128 loads with zero layout shuffling.

#define NN 4096
#define FF 256
#define HGAT 4
#define CC 64
#define HC 256
#define LEAKY 0.2f

typedef __attribute__((ext_vector_type(16))) _Float16 v16h;
typedef __attribute__((ext_vector_type(8)))  _Float16 v8h;
typedef __attribute__((ext_vector_type(8)))  float    v8f;
typedef __attribute__((ext_vector_type(4)))  float    f4;

// Fragment layout: blob[(blk*HGAT + h)*4 + cc][lane][slot]
//   blk = (k or j)/32, cc = col-chunk of 16, lane 0..31, slot 0..15 (halves).
// B element (k within 32-block, n = col within chunk) lives at
//   lane = n + 16*(k/16), slot = k%16   — the v16h B operand in the contiguous k order that A also uses.
__device__ __forceinline__ size_t frag_base_halves(int blk, int h, int cc) {
    return (size_t)(((blk * HGAT + h) * 4 + cc)) * 32 * 16;
}

// ---------------------------------------------------------------------------
// Kernel 0: pre-pack W [F, H*C] f32 -> f16 fragment order (128 KB, run once).
// tid decomposes exactly as frag_base_halves ordering.
// ---------------------------------------------------------------------------
__global__ __launch_bounds__(256) void gat_packw(
    const float* __restrict__ Wr, _Float16* __restrict__ wp16)
{
    const int tid  = blockIdx.x * 256 + threadIdx.x;   // 0 .. 65535
    const int slot = tid & 15;
    const int lane = (tid >> 4) & 31;
    const int cc   = (tid >> 9) & 3;
    const int h    = (tid >> 11) & 3;
    const int kb   = tid >> 13;                        // 0..7
    const int k    = kb * 32 + 16 * (lane >> 4) + slot;
    const int col  = h * CC + cc * 16 + (lane & 15);
    wp16[tid] = (_Float16)Wr[k * HC + col];
}

// ---------------------------------------------------------------------------
// Kernel 1: xp = x @ W (f16 WMMA, f32 accum). Writes xp16 in fragment order
// (one packed v8h store per chunk) and attention scalars s[n,h], nh[h][n].
// Block = 128 threads = 4 waves; wave w == head w; 16 rows per block.
// ---------------------------------------------------------------------------
__global__ __launch_bounds__(128) void gat_proj(
    const float* __restrict__ x, const _Float16* __restrict__ wp16,
    const float* __restrict__ aks, const float* __restrict__ akn,
    _Float16* __restrict__ xp16, float* __restrict__ sbuf,
    float* __restrict__ nh)
{
    const int i0   = blockIdx.x * 16;
    const int lane = threadIdx.x & 31;
    const int w    = threadIdx.x >> 5;   // wave id == head id
    const int g    = lane >> 4;          // half-wave group
    const int ln   = lane & 15;

    v8f acc[4] = {};
    const float* xrow = x + (size_t)(i0 + ln) * FF;  // A rows = lane%16

    for (int kb = 0; kb < FF / 32; ++kb) {
        const int k0 = kb * 32;
        // A f16 16x32: slot e -> k = 16g + e: one 16-float run.
        f4 x0 = *(const f4*)(xrow + k0 + 16 * g);
        f4 x1 = *(const f4*)(xrow + k0 + 16 * g + 4);
        f4 x2 = *(const f4*)(xrow + k0 + 16 * g + 8);
        f4 x3 = *(const f4*)(xrow + k0 + 16 * g + 12);
        v16h afrag;
        afrag[0]=(_Float16)x0.x;  afrag[1]=(_Float16)x0.y;
        afrag[2]=(_Float16)x0.z;  afrag[3]=(_Float16)x0.w;
        afrag[4]=(_Float16)x1.x;  afrag[5]=(_Float16)x1.y;
        afrag[6]=(_Float16)x1.z;  afrag[7]=(_Float16)x1.w;
        afrag[8]=(_Float16)x2.x;  afrag[9]=(_Float16)x2.y;
        afrag[10]=(_Float16)x2.z; afrag[11]=(_Float16)x2.w;
        afrag[12]=(_Float16)x3.x; afrag[13]=(_Float16)x3.y;
        afrag[14]=(_Float16)x3.z; afrag[15]=(_Float16)x3.w;

        // B fragments pre-packed: one 32B aligned load each (2x b128).
        const _Float16* wb = wp16 + frag_base_halves(kb, w, 0)
                                  + (size_t)lane * 16;
        #pragma unroll
        for (int cc = 0; cc < 4; ++cc) {
            v16h bfrag = *(const v16h*)(wb + (size_t)cc * (32 * 16));
            acc[cc] = __builtin_amdgcn_wmma_f32_16x16x32_f16(
                false, afrag, false, bfrag, (short)0, acc[cc], false, false);
        }
    }

    // Fragment-order store of xp. Rows i0..i0+15 are one half of 32-block
    // jb = i0>>5. For acc element r: k = (i0&16)+r+8g, so dest lane =
    // ln + 16*((i0>>4)&1) (const) and slots 8g..8g+7 (contiguous, 16B).
    {
        const int jb       = i0 >> 5;
        const int lane_dst = ln + 16 * ((i0 >> 4) & 1);
        #pragma unroll
        for (int cc = 0; cc < 4; ++cc) {
            v8h p;
            #pragma unroll
            for (int r = 0; r < 8; ++r) p[r] = (_Float16)acc[cc][r];
            _Float16* dst = xp16 + frag_base_halves(jb, w, cc)
                                 + (size_t)lane_dst * 16 + 8 * g;
            *(v8h*)dst = p;   // one 16B store per chunk
        }
    }

    // s[n,h] = sum_c xp[n,h,c]*aks[c,h]; nh[h][n] likewise with akn.
    // aks/akn are [C,H,1] row-major: (c,h) at c*HGAT + h.
    #pragma unroll
    for (int r = 0; r < 8; ++r) {
        float sp = 0.f, np = 0.f;
        #pragma unroll
        for (int cc = 0; cc < 4; ++cc) {
            int c = cc * 16 + ln;
            sp += acc[cc][r] * aks[c * HGAT + w];
            np += acc[cc][r] * akn[c * HGAT + w];
        }
        #pragma unroll
        for (int d = 1; d < 16; d <<= 1) {
            sp += __shfl_xor(sp, d, 32);
            np += __shfl_xor(np, d, 32);
        }
        if (ln == r) {
            int row = i0 + r + 8 * g;
            sbuf[row * HGAT + w] = sp;
            nh[w * NN + row]     = np;   // head-major for vectorized reads
        }
    }
}

// ---------------------------------------------------------------------------
// Kernel 2: nmax[h] = max_j n[j,h]  (softmax bound: leakyrelu is monotone,
// so M_i = leakyrelu(s_i + nmax_h) upper-bounds every logit in row i).
// ---------------------------------------------------------------------------
__global__ __launch_bounds__(128) void gat_nmax(
    const float* __restrict__ nh, float* __restrict__ nmax)
{
    const int h = threadIdx.x >> 5;
    const int lane = threadIdx.x & 31;
    float m = -3.0e38f;
    for (int j = lane; j < NN; j += 32) m = fmaxf(m, nh[h * NN + j]);
    #pragma unroll
    for (int d = 1; d < 32; d <<= 1) m = fmaxf(m, __shfl_xor(m, d, 32));
    if (lane == 0) nmax[h] = m;
}

// ---------------------------------------------------------------------------
// Kernel 3: fused masked softmax + P @ xp.
// Block = 128 threads = 4 waves; wave = head; 16 output rows per block.
// Per 32-wide j-tile: P built straight into WMMA A layout (16 v_exp_f32 per
// lane, TRANS ops co-execute with WMMA), B fragments are two b128 loads each.
// Adjacency rows stream sequentially; prefetch ~1KB ahead.
// ---------------------------------------------------------------------------
__global__ __launch_bounds__(128) void gat_attn(
    const float* __restrict__ a, const _Float16* __restrict__ xp16,
    const float* __restrict__ sbuf, const float* __restrict__ nh,
    const float* __restrict__ nmax, const float* __restrict__ bias,
    float* __restrict__ out)
{
    const int i0   = blockIdx.x * 16;
    const int lane = threadIdx.x & 31;
    const int h    = threadIdx.x >> 5;
    const int g    = lane >> 4;
    const int ln   = lane & 15;
    const int rowg = i0 + ln;                 // row this lane builds P for

    const float sm = sbuf[rowg * HGAT + h];
    float t = sm + nmax[h];
    const float Mrow = (t > 0.f) ? t : LEAKY * t;

    v8f  acc[4] = {};
    float Z = 0.f;
    const float* arow = a  + (size_t)rowg * NN;
    const float* nrow = nh + (size_t)h * NN;

    for (int j0 = 0; j0 < NN; j0 += 32) {
        // Prefetch the adjacency stream ~8 tiles (1KB) ahead.
        if (j0 + 256 < NN) __builtin_prefetch(arow + j0 + 256, 0, 0);

        // This lane's 16 columns: j0 + {16g..16g+15}
        f4 a0 = *(const f4*)(arow + j0 + 16 * g);
        f4 a1 = *(const f4*)(arow + j0 + 16 * g + 4);
        f4 a2 = *(const f4*)(arow + j0 + 16 * g + 8);
        f4 a3 = *(const f4*)(arow + j0 + 16 * g + 12);
        f4 n0 = *(const f4*)(nrow + j0 + 16 * g);
        f4 n1 = *(const f4*)(nrow + j0 + 16 * g + 4);
        f4 n2 = *(const f4*)(nrow + j0 + 16 * g + 8);
        f4 n3 = *(const f4*)(nrow + j0 + 16 * g + 12);
        float av[16], nv[16];
        av[0]=a0.x; av[1]=a0.y; av[2]=a0.z; av[3]=a0.w;
        av[4]=a1.x; av[5]=a1.y; av[6]=a1.z; av[7]=a1.w;
        av[8]=a2.x; av[9]=a2.y; av[10]=a2.z; av[11]=a2.w;
        av[12]=a3.x; av[13]=a3.y; av[14]=a3.z; av[15]=a3.w;
        nv[0]=n0.x; nv[1]=n0.y; nv[2]=n0.z; nv[3]=n0.w;
        nv[4]=n1.x; nv[5]=n1.y; nv[6]=n1.z; nv[7]=n1.w;
        nv[8]=n2.x; nv[9]=n2.y; nv[10]=n2.z; nv[11]=n2.w;
        nv[12]=n3.x; nv[13]=n3.y; nv[14]=n3.z; nv[15]=n3.w;

        v16h P;
        #pragma unroll
        for (int e = 0; e < 16; ++e) {
            int k   = 16 * g + e;
            int col = j0 + k;
            float l = sm + nv[e];
            l = (l > 0.f) ? l : LEAKY * l;
            bool keep = (av[e] != 0.f) || (col == rowg);  // self-loop on diag
            float p = keep ? __expf(l - Mrow) : 0.f;
            _Float16 ph = (_Float16)p;
            P[e] = ph;
            Z += (float)ph;   // normalize with the same (f16-rounded) values
        }

        // B fragments already laid out in operand order: 32B load per chunk.
        const int jb = j0 >> 5;
        const _Float16* xb = xp16 + frag_base_halves(jb, h, 0)
                                  + (size_t)lane * 16;
        #pragma unroll
        for (int cc = 0; cc < 4; ++cc) {
            v16h B = *(const v16h*)(xb + (size_t)cc * (32 * 16));
            acc[cc] = __builtin_amdgcn_wmma_f32_16x16x32_f16(
                false, P, false, B, (short)0, acc[cc], false, false);
        }
    }

    // Combine the two half-wave partial Z sums (k split across lane groups).
    Z += __shfl_xor(Z, 16, 32);

    // Normalize, add bias, store. acc element r corresponds to row r + 8g.
    #pragma unroll
    for (int cc = 0; cc < 4; ++cc) {
        const int col = h * CC + cc * 16 + ln;
        const float bv = bias[col];
        #pragma unroll
        for (int r = 0; r < 8; ++r) {
            int m = r + 8 * g;
            float zr = __shfl(Z, m, 32);   // lane m holds row m's total
            out[(size_t)(i0 + m) * HC + col] = acc[cc][r] / zr + bv;
        }
    }
}

// ---------------------------------------------------------------------------
extern "C" void kernel_launch(void* const* d_in, const int* in_sizes, int n_in,
                              void* d_out, int out_size, void* d_ws, size_t ws_size,
                              hipStream_t stream) {
    const float* x    = (const float*)d_in[0];   // [N,F]
    const float* a    = (const float*)d_in[1];   // [N,N]
    const float* Wr   = (const float*)d_in[2];   // [F,H,C] == [F, H*C] row-major
    const float* aks  = (const float*)d_in[3];   // [C,H,1]
    const float* akn  = (const float*)d_in[4];   // [C,H,1]
    const float* bias = (const float*)d_in[5];   // [H*C]
    float* out = (float*)d_out;

    char* ws = (char*)d_ws;
    const size_t XP_BYTES = (size_t)NN * HC * sizeof(_Float16);       // 2 MB
    _Float16* xp16 = (_Float16*)ws;
    float*    sbuf  = (float*)(ws + XP_BYTES);                        // 64 KB
    float*    nhbuf = (float*)(ws + XP_BYTES + 65536);                // 64 KB
    float*    nmaxb = (float*)(ws + XP_BYTES + 131072);               // 256 B pad
    _Float16* wp16  = (_Float16*)(ws + XP_BYTES + 131072 + 256);      // 128 KB

    gat_packw<<<256, 256, 0, stream>>>(Wr, wp16);
    gat_proj<<<NN / 16, 128, 0, stream>>>(x, wp16, aks, akn, xp16, sbuf, nhbuf);
    gat_nmax<<<1, 128, 0, stream>>>(nhbuf, nmaxb);
    gat_attn<<<NN / 16, 128, 0, stream>>>(a, xp16, sbuf, nhbuf, nmaxb, bias, out);
}
